// PTB_Model_6648609375046
// MI455X (gfx1250) — matrix operand order repaired and checked by reading, not compiled, not run on hardware
//
#include <hip/hip_runtime.h>
#include <hip/hip_bf16.h>

typedef __attribute__((ext_vector_type(16))) _Float16 v16h;
typedef __attribute__((ext_vector_type(8)))  float    v8f;

#define GEMM_WAVES 4
#define PW_THREADS 256

// Fragment geometry helpers ------------------------------------------------
// A-tile (16x32 f16): lane l = khalf*16 + m ; lane's 16 halfs cover
//   koff = khalf*8 + (e&7) + ((e>>3)<<4)  (e = element index 0..15)
// B-tile (32x16 f16): lane l = khalf*16 + n ; lane's 16 halfs cover
//   k = khalf*8 + (e&7) + ((e>>3)<<4)  (same order as A)
// Both stored lane-major: tile*512 + lane*16 + e  -> one 32B load per lane.

__device__ __forceinline__ v8f wmma_f16(v16h a, v16h b, v8f c) {
    return __builtin_amdgcn_wmma_f32_16x16x32_f16(false, a, false, b, (short)0, c, false, false);
}

// ---------------------------------------------------------------------------
// Weight packer: W [Ktot x N] row-major f32 -> f16 WMMA-B fragments,
// rows [row0, row0+K) zero-padded to ktiles*32. Tile (nt,kt) at
// ((nt*ktiles + kt)*512).
// ---------------------------------------------------------------------------
__global__ void __launch_bounds__(256)
pack_w_k(const float* __restrict__ W, int row0, int K, int ktiles, int ldw,
         _Float16* __restrict__ out, int total) {
    int idx = blockIdx.x * 256 + threadIdx.x;
    if (idx >= total) return;
    int e    = idx & 15;
    int lane = (idx >> 4) & 31;
    int tile = idx >> 9;
    int kt = tile % ktiles;
    int nt = tile / ktiles;
    int n = nt * 16 + (lane & 15);
    int k = kt * 32 + (lane >> 4) * 8 + (e & 7) + ((e >> 3) << 4);
    float v = (k < K) ? W[(size_t)(row0 + k) * ldw + n] : 0.0f;
    out[idx] = (_Float16)v;
}

// ---------------------------------------------------------------------------
// Input packer: inputs [B=128][T=100][E=128] f32 -> per-step A-fragments.
// Slot t stride = 8 mtiles * 4 ktiles * 512 = 16384 halfs; tile (mtile,kt)
// at (mtile*4 + kt)*512.
// ---------------------------------------------------------------------------
__global__ void __launch_bounds__(256)
pack_x_k(const float* __restrict__ inputs, _Float16* __restrict__ out, int total) {
    int idx = blockIdx.x * 256 + threadIdx.x;
    if (idx >= total) return;
    int e     = idx & 15;
    int lane  = (idx >> 4) & 31;
    int rest  = idx >> 9;
    int kt    = rest & 3;
    int mtile = (rest >> 2) & 7;
    int t     = rest >> 5;
    int m     = lane & 15;
    int khalf = lane >> 4;
    int koff  = khalf * 8 + (e & 7) + ((e >> 3) << 4);
    int k     = kt * 32 + koff;
    int b     = mtile * 16 + m;
    out[idx] = (_Float16)inputs[((size_t)b * 100 + t) * 128 + k];
}

__global__ void __launch_bounds__(256)
zero_k(float* __restrict__ p, int n) {
    int i = blockIdx.x * 256 + threadIdx.x;
    if (i < n) p[i] = 0.0f;
}

// ---------------------------------------------------------------------------
// Z[128 x N] = A_x @ Wx + A_h @ Wh ; all operands pre-packed f16 fragments.
// One 16x16 C tile per wave, NT consecutive N-tiles. Inner loop is pure
// b128 loads + v_wmma (no conversions).
// ---------------------------------------------------------------------------
template <int NT>
__global__ void __launch_bounds__(32 * GEMM_WAVES)
gemm2f_k(const _Float16* __restrict__ Ax, int kxt,
         const _Float16* __restrict__ Ah, int kht,
         const _Float16* __restrict__ Wx, const _Float16* __restrict__ Wh,
         float* __restrict__ Z, int ldz, int ngroups) {
    int lane = threadIdx.x;
    int g    = blockIdx.x * GEMM_WAVES + threadIdx.y;
    if (g >= ngroups) return;                 // wave-uniform: EXEC stays all-ones
    int mtile = blockIdx.y;
    int khalf = lane >> 4;
    const _Float16* ax = Ax + ((size_t)mtile * kxt) * 512 + lane * 16;
    const _Float16* ah = Ah + ((size_t)mtile * kht) * 512 + lane * 16;
    int nt0 = g * NT;

    v8f acc[NT] = {};
    for (int kt = 0; kt < kxt; ++kt) {
        v16h a = *(const v16h*)(ax + (size_t)kt * 512);
#pragma unroll
        for (int t = 0; t < NT; ++t) {
            v16h b = *(const v16h*)(Wx + ((size_t)(nt0 + t) * kxt + kt) * 512 + lane * 16);
            acc[t] = wmma_f16(a, b, acc[t]);
        }
    }
    for (int kt = 0; kt < kht; ++kt) {
        v16h a = *(const v16h*)(ah + (size_t)kt * 512);
#pragma unroll
        for (int t = 0; t < NT; ++t) {
            v16h b = *(const v16h*)(Wh + ((size_t)(nt0 + t) * kht + kt) * 512 + lane * 16);
            acc[t] = wmma_f16(a, b, acc[t]);
        }
    }
    // C layout: VGPR r, lane l -> m = (l>>4)*8 + r, n = l&15
#pragma unroll
    for (int t = 0; t < NT; ++t) {
        int col = (nt0 + t) * 16 + (lane & 15);
#pragma unroll
        for (int r = 0; r < 8; ++r) {
            int row = mtile * 16 + khalf * 8 + r;
            Z[(size_t)row * ldz + col] = acc[t][r];
        }
    }
}

// ---------------------------------------------------------------------------
// Output projection, f16-fragment A with the reference's row permutation:
// flat row r -> t = r%100, b = r/100, activation = history slot (t+1), row b.
// Each lane derives its own fragment base pointer from (t, b).
// Hf slot stride = 8*22*512 = 90112 halfs.
// ---------------------------------------------------------------------------
template <int NT>
__global__ void __launch_bounds__(32 * GEMM_WAVES)
gemm_out_k(const _Float16* __restrict__ Hf, const _Float16* __restrict__ W,
           const float* __restrict__ bias, float* __restrict__ out,
           int ktot, int ngroups) {
    int lane  = threadIdx.x;
    int g     = blockIdx.x * GEMM_WAVES + threadIdx.y;
    if (g >= ngroups) return;
    int mtile = blockIdx.y;
    int m     = lane & 15;
    int khalf = lane >> 4;
    int r = mtile * 16 + m;            // flat row 0..12799
    int t = r % 100;
    int b = r / 100;
    const _Float16* arow = Hf + (size_t)(t + 1) * 90112
                              + ((size_t)(b >> 4) * ktot) * 512
                              + ((khalf << 4) + (b & 15)) * 16;
    int nt0 = g * NT;

    v8f acc[NT] = {};
    for (int kt = 0; kt < ktot; ++kt) {
        v16h a = *(const v16h*)(arow + (size_t)kt * 512);
#pragma unroll
        for (int tt = 0; tt < NT; ++tt) {
            v16h bf = *(const v16h*)(W + ((size_t)(nt0 + tt) * ktot + kt) * 512 + lane * 16);
            acc[tt] = wmma_f16(a, bf, acc[tt]);
        }
    }
#pragma unroll
    for (int tt = 0; tt < NT; ++tt) {
        int col = (nt0 + tt) * 16 + (lane & 15);
        float bv = bias[col];
#pragma unroll
        for (int rr = 0; rr < 8; ++rr) {
            int row = mtile * 16 + khalf * 8 + rr;
            out[(size_t)row * 10000 + col] = acc[tt][rr] + bv;
        }
    }
}

// ---------------------------------------------------------------------------
// Fused LN4 + LSTM gates + cell-state LN + zoneout. One block per batch row.
// Writes f32 h (elementwise use) and the f16 A-fragment image (GEMM use,
// zero-padded K columns).
// ---------------------------------------------------------------------------
__device__ __forceinline__ float block_reduce(float v, float* red) {
    int tid = threadIdx.x;
    red[tid] = v;
    __syncthreads();
    for (int s = PW_THREADS / 2; s > 0; s >>= 1) {
        if (tid < s) red[tid] += red[tid + s];
        __syncthreads();
    }
    float r = red[0];
    __syncthreads();
    return r;
}

__device__ __forceinline__ float sigmoidf(float x) { return 1.0f / (1.0f + __expf(-x)); }

__device__ __forceinline__ void frag_store(_Float16* __restrict__ frag, int frag_ktiles,
                                           int b, int e, float val) {
    int mtile = b >> 4, m = b & 15;
    int kt = e >> 5, kr = e & 31;
    int khalf = (kr >> 3) & 1;
    int ee = (kr & 7) + ((kr >> 4) << 3);
    frag[((size_t)mtile * frag_ktiles + kt) * 512 + ((khalf << 4) + m) * 16 + ee] = (_Float16)val;
}

__global__ void __launch_bounds__(PW_THREADS)
lnlstm_k(const float* __restrict__ Z, const float* __restrict__ bz,
         const float* __restrict__ gg, const float* __restrict__ bg,
         const float* __restrict__ gc, const float* __restrict__ bc,
         float* __restrict__ cbuf,
         const float* hold, float* hout,                 // f32, stride hu (may alias)
         _Float16* __restrict__ hfrag, int frag_ktiles,  // f16 A-fragment image
         int hu, int hu_pad, float keep_h, float keep_c) {
    __shared__ float red[PW_THREADS];
    __shared__ float stat[10];   // mu[0..3], rs[0..3], mu_c, rs_c
    const float EPS = 1e-5f;
    int b   = blockIdx.x;
    int tid = threadIdx.x;
    const float* z = Z + (size_t)b * (4 * hu);
    float inv_hu = 1.0f / (float)hu;

    // Per-gate LN statistics
    for (int g = 0; g < 4; ++g) {
        float s = 0.0f, ss = 0.0f;
        for (int e = tid; e < hu; e += PW_THREADS) {
            float v = z[g * hu + e] + bz[g * hu + e];
            s += v; ss += v * v;
        }
        s  = block_reduce(s, red);
        ss = block_reduce(ss, red);
        if (tid == 0) {
            float mu = s * inv_hu;
            stat[g]     = mu;
            stat[4 + g] = rsqrtf(ss * inv_hu - mu * mu + EPS);
        }
    }
    __syncthreads();

    // Gate math + new cell state
    float ncl[3], ol[3], cold[3];
    float s = 0.0f, ss = 0.0f;
#pragma unroll
    for (int j = 0; j < 3; ++j) {
        int e = tid + j * PW_THREADS;
        ncl[j] = 0.0f; ol[j] = 0.0f; cold[j] = 0.0f;
        if (e < hu) {
            float zi = (z[0 * hu + e] + bz[0 * hu + e] - stat[0]) * stat[4] * gg[0 * hu + e] + bg[0 * hu + e];
            float zj = (z[1 * hu + e] + bz[1 * hu + e] - stat[1]) * stat[5] * gg[1 * hu + e] + bg[1 * hu + e];
            float zf = (z[2 * hu + e] + bz[2 * hu + e] - stat[2]) * stat[6] * gg[2 * hu + e] + bg[2 * hu + e];
            float zo = (z[3 * hu + e] + bz[3 * hu + e] - stat[3]) * stat[7] * gg[3 * hu + e] + bg[3 * hu + e];
            float cp = cbuf[(size_t)b * hu + e];
            float nc = cp * sigmoidf(zf + 1.0f) + sigmoidf(zi) * tanhf(zj);
            ncl[j] = nc; ol[j] = zo; cold[j] = cp;
            s += nc; ss += nc * nc;
        }
    }
    s  = block_reduce(s, red);
    ss = block_reduce(ss, red);
    if (tid == 0) {
        float mu = s * inv_hu;
        stat[8] = mu;
        stat[9] = rsqrtf(ss * inv_hu - mu * mu + EPS);
    }
    __syncthreads();

    // Hidden state + zoneout; dual-format output
#pragma unroll
    for (int j = 0; j < 3; ++j) {
        int e = tid + j * PW_THREADS;
        if (e < hu) {
            float nc = ncl[j];
            float nh = tanhf((nc - stat[8]) * stat[9] * gc[e] + bc[e]) * sigmoidf(ol[j]);
            float ho = hold[(size_t)b * hu + e];
            float hv = keep_h * nh + (1.0f - keep_h) * ho;
            hout[(size_t)b * hu + e] = hv;
            cbuf[(size_t)b * hu + e] = keep_c * nc + (1.0f - keep_c) * cold[j];
            frag_store(hfrag, frag_ktiles, b, e, hv);
        }
    }
    // zero K-pad columns in the fragment image
    for (int e = hu + tid; e < hu_pad; e += PW_THREADS)
        frag_store(hfrag, frag_ktiles, b, e, 0.0f);
}

// Final states -> tail of d_out: fh[128*700], fc[128*700], sh[128*400], sc[128*400]
__global__ void __launch_bounds__(256)
copy_states_k(const float* __restrict__ fh, const float* __restrict__ fc,
              const float* __restrict__ sh, const float* __restrict__ sc,
              float* __restrict__ out) {
    const int NF = 128 * 700, NS = 128 * 400;
    int i = blockIdx.x * 256 + threadIdx.x;
    if (i < NF) { out[i] = fh[i]; return; }
    i -= NF;
    if (i < NF) { out[NF + i] = fc[i]; return; }
    i -= NF;
    if (i < NS) { out[2 * NF + i] = sh[i]; return; }
    i -= NS;
    if (i < NS) { out[2 * NF + NS + i] = sc[i]; return; }
}

// ---------------------------------------------------------------------------
extern "C" void kernel_launch(void* const* d_in, const int* in_sizes, int n_in,
                              void* d_out, int out_size, void* d_ws, size_t ws_size,
                              hipStream_t stream) {
    (void)in_sizes; (void)n_in; (void)out_size; (void)ws_size;
    const float* inputs = (const float*)d_in[0];
    const float* W0  = (const float*)d_in[1];
    const float* b0  = (const float*)d_in[2];
    const float* g0  = (const float*)d_in[3];
    const float* bg0 = (const float*)d_in[4];
    const float* gc0 = (const float*)d_in[5];
    const float* bc0 = (const float*)d_in[6];
    const float* W1  = (const float*)d_in[7];
    const float* b1  = (const float*)d_in[8];
    const float* g1  = (const float*)d_in[9];
    const float* bg1 = (const float*)d_in[10];
    const float* gc1 = (const float*)d_in[11];
    const float* bc1 = (const float*)d_in[12];
    const float* WS  = (const float*)d_in[13];
    const float* bS  = (const float*)d_in[14];
    const float* gS  = (const float*)d_in[15];
    const float* bgS = (const float*)d_in[16];
    const float* gcS = (const float*)d_in[17];
    const float* bcS = (const float*)d_in[18];
    const float* Wout = (const float*)d_in[19];
    const float* bout = (const float*)d_in[20];
    float* out = (float*)d_out;

    // Workspace carving (256B aligned)
    size_t off = 0;
    char* ws = (char*)d_ws;
    auto carve = [&](size_t elems, size_t esz) {
        void* p = ws + off;
        off += ((elems * esz + 255) & ~(size_t)255);
        return p;
    };
    _Float16* pW0x = (_Float16*)carve((size_t)4  * 175 * 512, 2);   // E=128  -> 4 ktiles
    _Float16* pW0h = (_Float16*)carve((size_t)22 * 175 * 512, 2);   // F=704  -> 22
    _Float16* pWSx = (_Float16*)carve((size_t)22 * 100 * 512, 2);
    _Float16* pWSh = (_Float16*)carve((size_t)13 * 100 * 512, 2);   // S=416 -> 13
    _Float16* pW1x = (_Float16*)carve((size_t)13 * 175 * 512, 2);
    _Float16* pW1h = (_Float16*)carve((size_t)22 * 175 * 512, 2);
    _Float16* pWo  = (_Float16*)carve((size_t)22 * 625 * 512, 2);
    _Float16* Xf   = (_Float16*)carve((size_t)100 * 16384, 2);       // inputs, A-frag
    _Float16* Hf   = (_Float16*)carve((size_t)101 * 90112, 2);       // fh history, A-frag
    _Float16* Af   = (_Float16*)carve((size_t)90112, 2);             // fhA, A-frag
    _Float16* Sf   = (_Float16*)carve((size_t)53248, 2);             // sh, A-frag (13 ktiles)
    float* fh32  = (float*)carve((size_t)128 * 700, 4);
    float* fhA32 = (float*)carve((size_t)128 * 700, 4);
    float* fcb   = (float*)carve((size_t)128 * 700, 4);
    float* sh32  = (float*)carve((size_t)128 * 400, 4);
    float* scb   = (float*)carve((size_t)128 * 400, 4);
    float* Zb    = (float*)carve((size_t)128 * 2800, 4);

    // Pack weights + inputs into fragment order (once per launch; deterministic)
    auto pack = [&](const float* W, int row0, int K, int ktiles, int ntiles, int ldw, _Float16* dst) {
        int total = ktiles * ntiles * 512;
        pack_w_k<<<(total + 255) / 256, 256, 0, stream>>>(W, row0, K, ktiles, ldw, dst, total);
    };
    pack(W0,   0, 128,  4, 175,  2800, pW0x);
    pack(W0, 128, 700, 22, 175,  2800, pW0h);
    pack(WS,   0, 700, 22, 100,  1600, pWSx);
    pack(WS, 700, 400, 13, 100,  1600, pWSh);
    pack(W1,   0, 400, 13, 175,  2800, pW1x);
    pack(W1, 400, 700, 22, 175,  2800, pW1h);
    pack(Wout, 0, 700, 22, 625, 10000, pWo);
    {
        int total = 100 * 16384;
        pack_x_k<<<(total + 255) / 256, 256, 0, stream>>>(inputs, Xf, total);
    }

    // Zero initial states (f16 fragment slot 0 + sh fragment, f32 states)
    auto zero = [&](void* p, int nfloats) {
        zero_k<<<(nfloats + 255) / 256, 256, 0, stream>>>((float*)p, nfloats);
    };
    zero(Hf,   90112 / 2);   // slot 0 (halfs -> float count)
    zero(Sf,   53248 / 2);
    zero(fh32, 128 * 700);
    zero(fcb,  128 * 700);
    zero(sh32, 128 * 400);
    zero(scb,  128 * 400);

    dim3 gblk(32, GEMM_WAVES);
    for (int t = 0; t < 100; ++t) {
        const _Float16* xt = Xf + (size_t)t * 16384;
        const _Float16* hp = Hf + (size_t)t * 90112;
        _Float16*       hn = Hf + (size_t)(t + 1) * 90112;

        // Fast cell 0: z = [x_t, fh] @ W0
        gemm2f_k<1><<<dim3(44, 8), gblk, 0, stream>>>(xt, 4, hp, 22,
                                                      pW0x, pW0h, Zb, 2800, 175);
        lnlstm_k<<<128, PW_THREADS, 0, stream>>>(Zb, b0, g0, bg0, gc0, bc0,
                                                 fcb, fh32, fhA32, Af, 22,
                                                 700, 704, 0.9f, 0.5f);
        // Slow cell: z = [fh, sh] @ WS
        gemm2f_k<1><<<dim3(25, 8), gblk, 0, stream>>>(Af, 22, Sf, 13,
                                                      pWSx, pWSh, Zb, 1600, 100);
        lnlstm_k<<<128, PW_THREADS, 0, stream>>>(Zb, bS, gS, bgS, gcS, bcS,
                                                 scb, sh32, sh32, Sf, 13,
                                                 400, 416, 0.9f, 0.5f);
        // Fast cell 1: z = [sh, fh] @ W1
        gemm2f_k<1><<<dim3(44, 8), gblk, 0, stream>>>(Sf, 13, Af, 22,
                                                      pW1x, pW1h, Zb, 2800, 175);
        lnlstm_k<<<128, PW_THREADS, 0, stream>>>(Zb, b1, g1, bg1, gc1, bc1,
                                                 fcb, fhA32, fh32, hn, 22,
                                                 700, 704, 0.9f, 0.5f);
    }

    // Output projection: 12800 x 10000, K=22 tiles, NT=5 (625/5 = 125 groups)
    gemm_out_k<5><<<dim3(32, 800), gblk, 0, stream>>>(Hf, pWo, bout, out, 22, 125);

    // Final carry states after the logits
    const int NSTATE = 128 * 700 * 2 + 128 * 400 * 2;
    copy_states_k<<<(NSTATE + 255) / 256, 256, 0, stream>>>(
        fh32, fcb, sh32, scb, out + 128000000LL);
}
